// ShapedTransformerLayer_50766513439142
// MI455X (gfx1250) — matrix operand order repaired and checked by reading, not compiled, not run on hardware
//
#include <hip/hip_runtime.h>
#include <hip/hip_bf16.h>
#include <math.h>

// ---------------------------------------------------------------------------
// ShapedTransformerLayer for MI455X (gfx1250), bf16 WMMA (v_wmma_f32_16x16x32_bf16)
// B=8, S=2048, D=512, DFF=2048.
// Pipelining style: two K-steps per iteration with fresh locals (no register
// rotation -> no v_mov storms), loads of both half-steps issued before WMMAs.
// ---------------------------------------------------------------------------

typedef __bf16 bf16_t;
typedef __attribute__((ext_vector_type(16))) __bf16 v16bf;
typedef __attribute__((ext_vector_type(8)))  __bf16 v8bf;
typedef __attribute__((ext_vector_type(8)))  float  v8f;

#define B_  8
#define S_  2048
#define D_  512
#define F_  2048
#define M_  (B_ * S_)          // 16384 total rows

#define WMMA_BF16(a, b, c) \
  __builtin_amdgcn_wmma_f32_16x16x32_bf16(false, (a), false, (b), (short)0, (c), false, false)

// ---- Fragment loaders (CDNA5 wave32 VGPR layouts, 16x16x32 bf16) ----------
// A matrix 16x32: lane m=lane&15 holds row m; lane-half h picks K ranges
__device__ __forceinline__ v16bf load_a_frag(const bf16_t* base, int ld) {
  const int lane = threadIdx.x & 31;
  const int m = lane & 15, h = lane >> 4;
  const bf16_t* rp = base + (size_t)m * ld + h * 8;
  v8bf lo = *(const v8bf*)(rp);        // K = h*8 .. h*8+7
  v8bf hi = *(const v8bf*)(rp + 16);   // K = 16+h*8 .. 16+h*8+7
  return __builtin_shufflevector(lo, hi, 0,1,2,3,4,5,6,7,8,9,10,11,12,13,14,15);
}

// B matrix 32x16 from "Bt" storage (Bt[n][k] row-major = B transposed):
// lane column n=lane&15, half h picks K = h*8 + slot (slot<8), 16 + h*8 + (slot-8)
__device__ __forceinline__ v16bf load_bt_frag(const bf16_t* base, int ld) {
  const int lane = threadIdx.x & 31;
  const int n = lane & 15, h = lane >> 4;
  const bf16_t* rp = base + (size_t)n * ld + h * 8;
  v8bf lo = *(const v8bf*)(rp);
  v8bf hi = *(const v8bf*)(rp + 16);
  return __builtin_shufflevector(lo, hi, 0,1,2,3,4,5,6,7,8,9,10,11,12,13,14,15);
}

// C/D f32 16x16: element r of a lane -> row (r + 8*(lane>>4)), col (lane&15)

// ---------------------------------------------------------------------------
__global__ void __launch_bounds__(256) cvt_bf16_kernel(bf16_t* __restrict__ dst,
                                                       const float* __restrict__ src,
                                                       int n) {
  int i = blockIdx.x * 256 + threadIdx.x;
  if (i < n) dst[i] = (bf16_t)src[i];
}

// ---------------------------------------------------------------------------
// Q/K projection: out[i,e] = sum_d srcb[i,d] * W[e,d] + bias[e].
// blockIdx.y: 0 -> q, 1 -> k. Block = 16 rows x 512 cols, 4 tiles/wave.
__global__ void __launch_bounds__(256) proj_kernel(
    const bf16_t* __restrict__ srcb,
    const bf16_t* __restrict__ Wqb, const float* __restrict__ bq, bf16_t* __restrict__ qo,
    const bf16_t* __restrict__ Wkb, const float* __restrict__ bk, bf16_t* __restrict__ ko) {
  const bf16_t* W   = blockIdx.y ? Wkb : Wqb;
  const float*  bia = blockIdx.y ? bk  : bq;
  bf16_t*       out = blockIdx.y ? ko  : qo;

  const int row0 = blockIdx.x * 16;
  const int wave = threadIdx.x >> 5;
  const int lane = threadIdx.x & 31;
  const bf16_t* arow = srcb + (size_t)row0 * D_;

  v8f acc[4] = {};
#pragma unroll
  for (int k0 = 0; k0 < D_; k0 += 64) {
    // both half-steps' fragments loaded before any WMMA issues
    v16bf a0 = load_a_frag(arow + k0, D_);
    v16bf b0[4];
#pragma unroll
    for (int t = 0; t < 4; ++t)
      b0[t] = load_bt_frag(W + (size_t)((wave * 4 + t) * 16) * D_ + k0, D_);
    v16bf a1 = load_a_frag(arow + k0 + 32, D_);
    v16bf b1[4];
#pragma unroll
    for (int t = 0; t < 4; ++t)
      b1[t] = load_bt_frag(W + (size_t)((wave * 4 + t) * 16) * D_ + k0 + 32, D_);
#pragma unroll
    for (int t = 0; t < 4; ++t) acc[t] = WMMA_BF16(a0, b0[t], acc[t]);
#pragma unroll
    for (int t = 0; t < 4; ++t) acc[t] = WMMA_BF16(a1, b1[t], acc[t]);
  }

  const int n = lane & 15, h = lane >> 4;
#pragma unroll
  for (int t = 0; t < 4; ++t) {
    int n0 = (wave * 4 + t) * 16;
    float bz = bia[n0 + n];
#pragma unroll
    for (int r = 0; r < 8; ++r) {
      int m = r + 8 * h;
      out[(size_t)(row0 + m) * D_ + n0 + n] = (bf16_t)(acc[t][r] + bz);
    }
  }
}

// ---------------------------------------------------------------------------
// LayerNorm over a 16x512 LDS tile, write f32 (and optional bf16) to global.
__device__ __forceinline__ void ln_tile_store(const float* xt, float* red,
                                              const float* __restrict__ w,
                                              const float* __restrict__ bb,
                                              float* out32, bf16_t* out16,
                                              size_t gbase) {
  const int tid = threadIdx.x;
  const int row = tid >> 4, sub = tid & 15;
  float s = 0.f, ss = 0.f;
#pragma unroll 4
  for (int i = 0; i < 32; ++i) {
    float v = xt[row * D_ + sub + i * 16];
    s += v; ss += v * v;
  }
  red[tid] = s; red[256 + tid] = ss;
  __syncthreads();
  if (sub == 0) {
    float S = 0.f, SS = 0.f;
    for (int j = 0; j < 16; ++j) { S += red[row * 16 + j]; SS += red[256 + row * 16 + j]; }
    float mu  = S * (1.0f / D_);
    float var = SS * (1.0f / D_) - mu * mu;
    red[512 + row] = mu;
    red[528 + row] = rsqrtf(var + 1e-5f);
  }
  __syncthreads();
  float mu = red[512 + row], rs = red[528 + row];
  for (int i = 0; i < 32; ++i) {
    int d = sub + i * 16;
    float v = (xt[row * D_ + d] - mu) * rs * w[d] + bb[d];
    if (out32) out32[gbase + (size_t)row * D_ + d] = v;
    if (out16) out16[gbase + (size_t)row * D_ + d] = (bf16_t)v;
  }
}

// ---------------------------------------------------------------------------
// Fused attention + residual + LN1. One block = 16 query rows of one batch.
// Dynamic LDS layout (bytes):
//   [0,131072)        scores 16x2048 f32   (reused as xt 16x512 f32 in epilogue)
//   [131072,196608)   attnb  16x2048 bf16
//   [196608,212992)   qt     16x512  bf16
//   [212992,294912)   srcT   2 x (512x40) bf16  (double buffer, padded ld=40)
//   [294912,297216)   red    576 f32
__global__ void __launch_bounds__(256) attn_kernel(
    const float* __restrict__ src, const bf16_t* __restrict__ srcb,
    const bf16_t* __restrict__ qb, const bf16_t* __restrict__ kb,
    const float* __restrict__ ln1w, const float* __restrict__ ln1b,
    float* __restrict__ xf, bf16_t* __restrict__ xb) {
  extern __shared__ char smem[];
  float*  scores = (float*)smem;
  bf16_t* attnb  = (bf16_t*)(smem + 131072);
  bf16_t* qt     = (bf16_t*)(smem + 196608);
  bf16_t* srcT0  = (bf16_t*)(smem + 212992);
  bf16_t* srcT1  = (bf16_t*)(smem + 212992 + 40960);
  float*  red    = (float*)(smem + 294912);
  float*  xt     = scores;  // reuse after phase B

  const int batch = blockIdx.y;
  const int row0  = blockIdx.x * 16;
  const int tid   = threadIdx.x;
  const int wave  = tid >> 5;
  const int lane  = tid & 31;
  const int n     = lane & 15, h = lane >> 4;
  const size_t rowg = (size_t)(batch * S_ + row0);  // global row index
  const float rsD = 0.044194173824159216f;          // 1/sqrt(512)

  // stage q tile (rows contiguous in global)
  for (int idx = tid; idx < 16 * D_; idx += 256) qt[idx] = qb[rowg * D_ + idx];
  __syncthreads();

  // ---- Phase A: scores = (q . k^T) / sqrt(D) ----
  const bf16_t* kbase = kb + (size_t)batch * S_ * D_;
  for (int g = 0; g < 4; ++g) {
    v8f acc[4] = {};
#pragma unroll
    for (int k0 = 0; k0 < D_; k0 += 64) {
      v16bf a0 = load_a_frag(qt + k0, D_);
      v16bf b0[4];
#pragma unroll
      for (int t = 0; t < 4; ++t)
        b0[t] = load_bt_frag(kbase + (size_t)((wave * 16 + g * 4 + t) * 16) * D_ + k0, D_);
      v16bf a1 = load_a_frag(qt + k0 + 32, D_);
      v16bf b1[4];
#pragma unroll
      for (int t = 0; t < 4; ++t)
        b1[t] = load_bt_frag(kbase + (size_t)((wave * 16 + g * 4 + t) * 16) * D_ + k0 + 32, D_);
#pragma unroll
      for (int t = 0; t < 4; ++t) acc[t] = WMMA_BF16(a0, b0[t], acc[t]);
#pragma unroll
      for (int t = 0; t < 4; ++t) acc[t] = WMMA_BF16(a1, b1[t], acc[t]);
    }
#pragma unroll
    for (int t = 0; t < 4; ++t) {
      int t0 = (wave * 16 + g * 4 + t) * 16;
#pragma unroll
      for (int r = 0; r < 8; ++r)
        scores[(r + 8 * h) * S_ + t0 + n] = acc[t][r] * rsD;
    }
  }
  __syncthreads();

  // ---- Phase B: softmax + eye - mean, write bf16 attn ----
  {
    const int row = tid >> 4, sub = tid & 15;
    float mx = -3.4e38f;
    for (int i = 0; i < 128; ++i) mx = fmaxf(mx, scores[row * S_ + sub + i * 16]);
    red[tid] = mx;
    __syncthreads();
    if (sub == 0) {
      float m = red[row * 16];
      for (int j = 1; j < 16; ++j) m = fmaxf(m, red[row * 16 + j]);
      red[256 + row] = m;
    }
    __syncthreads();
    float rmax = red[256 + row];
    float sum = 0.f;
    for (int i = 0; i < 128; ++i) {
      int j = sub + i * 16;
      float e = __expf(scores[row * S_ + j] - rmax);
      scores[row * S_ + j] = e;
      sum += e;
    }
    red[tid] = sum;
    __syncthreads();
    if (sub == 0) {
      float s2 = 0.f;
      for (int j = 0; j < 16; ++j) s2 += red[row * 16 + j];
      red[272 + row] = 1.0f / s2;
    }
    __syncthreads();
    float inv = red[272 + row];
    const float mean = 2.0f / (float)S_;  // (sum softmax + sum eye)/S
    const int sglobal = row0 + row;
    for (int i = 0; i < 128; ++i) {
      int j = sub + i * 16;
      float p = scores[row * S_ + j] * inv;
      if (j == sglobal) p += 1.0f;
      attnb[row * S_ + j] = (bf16_t)(p - mean);
    }
  }
  __syncthreads();

  // ---- Phase C: src2 = attn . src  (K=2048, double-buffered LDS panels) ----
  const bf16_t* sbase = srcb + (size_t)batch * S_ * D_;
  // prologue: stage panel for kb0 = 0 into srcT0
  for (int idx = tid; idx < 32 * D_; idx += 256) {
    int t = idx >> 9, d = idx & (D_ - 1);
    srcT0[d * 40 + t] = sbase[(size_t)t * D_ + d];
  }
  __syncthreads();

  v8f acc[4] = {};
  for (int it = 0; it < S_ / 32; ++it) {
    const int kb0 = it * 32;
    bf16_t* cur = (it & 1) ? srcT1 : srcT0;
    bf16_t* nxt = (it & 1) ? srcT0 : srcT1;
    if (it + 1 < S_ / 32) {
      // stage next panel (disjoint LDS region; overlaps with WMMAs below)
      for (int idx = tid; idx < 32 * D_; idx += 256) {
        int t = idx >> 9, d = idx & (D_ - 1);
        nxt[d * 40 + t] = sbase[(size_t)(kb0 + 32 + t) * D_ + d];
      }
    }
    v16bf a = load_a_frag(attnb + kb0, S_);
#pragma unroll
    for (int t = 0; t < 4; ++t) {
      int n0 = (wave * 4 + t) * 16;
      v16bf b = load_bt_frag(cur + (size_t)n0 * 40, 40);
      acc[t] = WMMA_BF16(a, b, acc[t]);
    }
    __syncthreads();  // next panel staged AND current panel consumed
  }

  // residual + stash in LDS tile (reuses scores region)
#pragma unroll
  for (int t = 0; t < 4; ++t) {
    int n0 = (wave * 4 + t) * 16;
#pragma unroll
    for (int r = 0; r < 8; ++r) {
      int m = r + 8 * h;
      xt[m * D_ + n0 + n] = acc[t][r] + src[(rowg + m) * D_ + n0 + n];
    }
  }
  __syncthreads();

  // LayerNorm1 -> x (f32 + bf16)
  ln_tile_store(xt, red, ln1w, ln1b, xf, xb, rowg * D_);
}

// ---------------------------------------------------------------------------
// FFN1: h = shaped_relu(x . W1^T + b1), bf16 out. Block = 16 rows x 256 cols.
__global__ void __launch_bounds__(256) ffn1_kernel(
    const bf16_t* __restrict__ xb, const bf16_t* __restrict__ W1b,
    const float* __restrict__ b1, bf16_t* __restrict__ hb) {
  const int row0 = blockIdx.x * 16;
  const int col0 = blockIdx.y * 256;
  const int wave = threadIdx.x >> 5;
  const int lane = threadIdx.x & 31;
  const bf16_t* arow = xb + (size_t)row0 * D_;
  const float c = 0.02209708691207961f;  // 1/sqrt(2048)

  v8f acc[2] = {};
#pragma unroll
  for (int k0 = 0; k0 < D_; k0 += 64) {
    v16bf a0 = load_a_frag(arow + k0, D_);
    v16bf b0[2];
#pragma unroll
    for (int t = 0; t < 2; ++t)
      b0[t] = load_bt_frag(W1b + (size_t)(col0 + (wave * 2 + t) * 16) * D_ + k0, D_);
    v16bf a1 = load_a_frag(arow + k0 + 32, D_);
    v16bf b1[2];
#pragma unroll
    for (int t = 0; t < 2; ++t)
      b1[t] = load_bt_frag(W1b + (size_t)(col0 + (wave * 2 + t) * 16) * D_ + k0 + 32, D_);
#pragma unroll
    for (int t = 0; t < 2; ++t) acc[t] = WMMA_BF16(a0, b0[t], acc[t]);
#pragma unroll
    for (int t = 0; t < 2; ++t) acc[t] = WMMA_BF16(a1, b1[t], acc[t]);
  }

  const int n = lane & 15, h = lane >> 4;
#pragma unroll
  for (int t = 0; t < 2; ++t) {
    int n0 = col0 + (wave * 2 + t) * 16;
    float bz = b1[n0 + n];
#pragma unroll
    for (int r = 0; r < 8; ++r) {
      int m = r + 8 * h;
      float v = acc[t][r] + bz;
      v = 1.0f + c * fmaxf(v, 0.0f) + (1.0f - c) * fminf(v, 0.0f);
      hb[(size_t)(row0 + m) * F_ + n0 + n] = (bf16_t)v;
    }
  }
}

// ---------------------------------------------------------------------------
// FFN2 + residual + LN2 -> out. Block = 16 rows x 512 cols (full row for LN).
__global__ void __launch_bounds__(256) ffn2_ln_kernel(
    const bf16_t* __restrict__ hb, const bf16_t* __restrict__ W2b,
    const float* __restrict__ b2, const float* __restrict__ xf,
    const float* __restrict__ ln2w, const float* __restrict__ ln2b,
    float* __restrict__ out) {
  __shared__ float xt[16 * D_];
  __shared__ float red[576];

  const int row0 = blockIdx.x * 16;
  const int wave = threadIdx.x >> 5;
  const int lane = threadIdx.x & 31;
  const bf16_t* arow = hb + (size_t)row0 * F_;

  v8f acc[4] = {};
  // K = 2048: rolled loop (loop-carried state = acc only; fresh locals inside)
  for (int k0 = 0; k0 < F_; k0 += 64) {
    v16bf a0 = load_a_frag(arow + k0, F_);
    v16bf b0[4];
#pragma unroll
    for (int t = 0; t < 4; ++t)
      b0[t] = load_bt_frag(W2b + (size_t)((wave * 4 + t) * 16) * F_ + k0, F_);
    v16bf a1 = load_a_frag(arow + k0 + 32, F_);
    __builtin_prefetch(arow + k0 + 64, 0, 1);  // global_prefetch upcoming A panel
    v16bf b1[4];
#pragma unroll
    for (int t = 0; t < 4; ++t)
      b1[t] = load_bt_frag(W2b + (size_t)((wave * 4 + t) * 16) * F_ + k0 + 32, F_);
#pragma unroll
    for (int t = 0; t < 4; ++t) acc[t] = WMMA_BF16(a0, b0[t], acc[t]);
#pragma unroll
    for (int t = 0; t < 4; ++t) acc[t] = WMMA_BF16(a1, b1[t], acc[t]);
  }

  const int n = lane & 15, h = lane >> 4;
#pragma unroll
  for (int t = 0; t < 4; ++t) {
    int n0 = (wave * 4 + t) * 16;
    float bz = b2[n0 + n];
#pragma unroll
    for (int r = 0; r < 8; ++r) {
      int m = r + 8 * h;
      xt[m * D_ + n0 + n] = acc[t][r] + bz + xf[(size_t)(row0 + m) * D_ + n0 + n];
    }
  }
  __syncthreads();
  ln_tile_store(xt, red, ln2w, ln2b, out, nullptr, (size_t)row0 * D_);
}

// ---------------------------------------------------------------------------
extern "C" void kernel_launch(void* const* d_in, const int* in_sizes, int n_in,
                              void* d_out, int out_size, void* d_ws, size_t ws_size,
                              hipStream_t stream) {
  const float* src  = (const float*)d_in[0];
  const float* Wq   = (const float*)d_in[1];
  const float* bq   = (const float*)d_in[2];
  const float* Wk   = (const float*)d_in[3];
  const float* bk   = (const float*)d_in[4];
  const float* W1   = (const float*)d_in[5];
  const float* b1   = (const float*)d_in[6];
  const float* W2   = (const float*)d_in[7];
  const float* b2   = (const float*)d_in[8];
  const float* ln1w = (const float*)d_in[9];
  const float* ln1b = (const float*)d_in[10];
  const float* ln2w = (const float*)d_in[11];
  const float* ln2b = (const float*)d_in[12];
  float* out = (float*)d_out;

  char* ws = (char*)d_ws;
  size_t off = 0;
  bf16_t* srcb = (bf16_t*)(ws + off); off += (size_t)M_ * D_ * 2;   // 16 MB
  bf16_t* qb   = (bf16_t*)(ws + off); off += (size_t)M_ * D_ * 2;
  bf16_t* kb   = (bf16_t*)(ws + off); off += (size_t)M_ * D_ * 2;
  bf16_t* Wqb  = (bf16_t*)(ws + off); off += (size_t)D_ * D_ * 2;
  bf16_t* Wkb  = (bf16_t*)(ws + off); off += (size_t)D_ * D_ * 2;
  bf16_t* W1b  = (bf16_t*)(ws + off); off += (size_t)F_ * D_ * 2;
  bf16_t* W2b  = (bf16_t*)(ws + off); off += (size_t)D_ * F_ * 2;
  float*  xf   = (float*)(ws + off);  off += (size_t)M_ * D_ * 4;   // 32 MB
  bf16_t* xb   = (bf16_t*)(ws + off); off += (size_t)M_ * D_ * 2;
  bf16_t* hb   = (bf16_t*)(ws + off); off += (size_t)M_ * F_ * 2;   // 64 MB

  // 1) bf16 conversions
  cvt_bf16_kernel<<<(M_ * D_ + 255) / 256, 256, 0, stream>>>(srcb, src, M_ * D_);
  cvt_bf16_kernel<<<(D_ * D_ + 255) / 256, 256, 0, stream>>>(Wqb, Wq, D_ * D_);
  cvt_bf16_kernel<<<(D_ * D_ + 255) / 256, 256, 0, stream>>>(Wkb, Wk, D_ * D_);
  cvt_bf16_kernel<<<(F_ * D_ + 255) / 256, 256, 0, stream>>>(W1b, W1, F_ * D_);
  cvt_bf16_kernel<<<(D_ * F_ + 255) / 256, 256, 0, stream>>>(W2b, W2, D_ * F_);

  // 2) Q/K projections
  proj_kernel<<<dim3(M_ / 16, 2), 256, 0, stream>>>(srcb, Wqb, bq, qb, Wkb, bk, kb);

  // 3) fused attention + residual + LN1
  const size_t attn_lds = 297216;  // see layout in attn_kernel
  attn_kernel<<<dim3(S_ / 16, B_), 256, attn_lds, stream>>>(src, srcb, qb, kb,
                                                            ln1w, ln1b, xf, xb);

  // 4) FFN1 + shaped ReLU
  ffn1_kernel<<<dim3(M_ / 16, F_ / 256), 256, 0, stream>>>(xb, W1b, b1, hb);

  // 5) FFN2 + residual + LN2 -> out
  ffn2_ln_kernel<<<dim3(M_ / 16), 256, 0, stream>>>(hb, W2b, b2, xf, ln2w, ln2b, out);
}
